// ScaledDotProductAttention_18133351923718
// MI455X (gfx1250) — matrix operand order repaired and checked by reading, not compiled, not run on hardware
//
#include <hip/hip_runtime.h>

// Scaled dot-product attention (flash-attention style) for MI455X / gfx1250.
// B=4, H=16, S=2048, D=64, fp32 in/out, key_padding_mask is jax bool (1 byte).
// Math done in f16 WMMA (v_wmma_f32_16x16x32_f16) with f32 accumulation.
// BK=64 keys per iteration: 16 WMMAs/iter, softmax reduction cost amortized.

typedef __attribute__((ext_vector_type(16))) _Float16 v16h;
typedef __attribute__((ext_vector_type(8)))  _Float16 v8h;
typedef __attribute__((ext_vector_type(8)))  float    v8f;
typedef __attribute__((ext_vector_type(4)))  float    v4f;

#define ATT_B 4
#define ATT_H 16
#define ATT_S 2048
#define ATT_D 64
#define BQ 64      // queries per block (4 waves x 16)
#define BK 64      // keys per iteration (two K=32 PV chunks)
#define NEGBIG (-1.0e9f)

__device__ __forceinline__ float redmax16(float v) {
    v = fmaxf(v, __shfl_xor(v, 1, 32));
    v = fmaxf(v, __shfl_xor(v, 2, 32));
    v = fmaxf(v, __shfl_xor(v, 4, 32));
    v = fmaxf(v, __shfl_xor(v, 8, 32));
    return v;
}
__device__ __forceinline__ float redsum16(float v) {
    v += __shfl_xor(v, 1, 32);
    v += __shfl_xor(v, 2, 32);
    v += __shfl_xor(v, 4, 32);
    v += __shfl_xor(v, 8, 32);
    return v;
}

__device__ __forceinline__ v16h pack16(const _Float16* p) {
    v8h lo = *(const v8h*)(p);
    v8h hi = *(const v8h*)(p + 16);
    return __builtin_shufflevector(lo, hi,
        0,1,2,3,4,5,6,7,8,9,10,11,12,13,14,15);
}

__global__ __launch_bounds__(128) void sdpa_wmma_kernel(
    const float* __restrict__ Q, const float* __restrict__ K,
    const float* __restrict__ V, const unsigned char* __restrict__ mask,
    float* __restrict__ O)
{
    __shared__ _Float16 Kh[BK * ATT_D];      // [key][d]   row-major f16  (8KB)
    __shared__ _Float16 Vt[ATT_D * BK];      // [d][key]   transposed f16 (8KB)
    __shared__ _Float16 Ps[4][16 * BK];      // per-wave P scratch [row][key] (8KB)

    const int tid  = threadIdx.x;
    const int wave = tid >> 5;
    const int lane = tid & 31;
    const int grp  = lane >> 4;      // 0 = lanes 0-15, 1 = lanes 16-31
    const int lm   = lane & 15;

    const int bh = blockIdx.y;                   // b*H + h
    const int b  = bh / ATT_H;
    const int q0 = blockIdx.x * BQ + wave * 16;  // this wave's 16 query rows

    const float scale = 0.125f;                  // 1/sqrt(64)

    // ---- Load Q tile (16x64) into WMMA A-matrix layout: two K=32 chunks.
    // A layout (16-bit, 16x32): lane = M (=lm); halves 0-7 -> K = 8*grp+0..7,
    // halves 8-15 -> K = 16+8*grp+0..7 (per ISA 7.12.2 table).
    v16h a[2];
    {
        const float* qrow = Q + ((size_t)bh * ATT_S + q0 + lm) * ATT_D;
        #pragma unroll
        for (int c = 0; c < 2; ++c) {
            #pragma unroll
            for (int seg = 0; seg < 2; ++seg) {
                const int d0 = c * 32 + seg * 16 + grp * 8;
                v4f f0 = *(const v4f*)(qrow + d0);
                v4f f1 = *(const v4f*)(qrow + d0 + 4);
                #pragma unroll
                for (int i = 0; i < 4; ++i) {
                    a[c][seg * 8 + i]     = (_Float16)(f0[i] * scale);
                    a[c][seg * 8 + 4 + i] = (_Float16)(f1[i] * scale);
                }
            }
        }
    }

    // Running state. C/D layout: lane holds column N=lm, rows M = r + 8*grp.
    v8f acc[4];
    float mrow[8], lrow[8];
    #pragma unroll
    for (int d = 0; d < 4; ++d) {
        v8f z = {0.f,0.f,0.f,0.f,0.f,0.f,0.f,0.f};
        acc[d] = z;
    }
    #pragma unroll
    for (int r = 0; r < 8; ++r) { mrow[r] = -1.0e30f; lrow[r] = 0.0f; }

    for (int kb = 0; kb < ATT_S; kb += BK) {
        // ---- cooperative stage: K (row major) and V (transposed) -> LDS f16
        // 64 rows x 64 cols of each; 128 threads -> 32 floats of K + V each.
        {
            const int row = tid >> 1;          // 0..63
            const int col = (tid & 1) * 32;    // 0 or 32
            const float* kr = K + ((size_t)bh * ATT_S + kb + row) * ATT_D + col;
            const float* vr = V + ((size_t)bh * ATT_S + kb + row) * ATT_D + col;
            if (kb + BK < ATT_S) {   // prefetch next tile into cache
                __builtin_prefetch((const void*)(kr + (size_t)BK * ATT_D), 0, 0);
                __builtin_prefetch((const void*)(vr + (size_t)BK * ATT_D), 0, 0);
            }
            #pragma unroll
            for (int i = 0; i < 8; ++i) {
                v4f kf = *(const v4f*)(kr + 4 * i);
                v4f vf = *(const v4f*)(vr + 4 * i);
                #pragma unroll
                for (int j = 0; j < 4; ++j) {
                    Kh[row * ATT_D + col + 4 * i + j] = (_Float16)kf[j];
                    Vt[(col + 4 * i + j) * BK + row] = (_Float16)vf[j];
                }
            }
        }
        __syncthreads();

        // ---- scores: four 16x16 tiles (keys kb+n*16 .. +15), contraction D=64
        // B layout (16-bit, 32x16): lane col N=lm (=key), halves e<8 -> K row =
        // grp*8 + e, e>=8 -> 16+grp*8+(e-8) (= local d). B[d][key] = K[key][d] in Kh.
        v8f sc[4];
        #pragma unroll
        for (int n = 0; n < 4; ++n) {
            v8f cacc = {0.f,0.f,0.f,0.f,0.f,0.f,0.f,0.f};
            #pragma unroll
            for (int c = 0; c < 2; ++c) {
                v16h bt = pack16(&Kh[(n * 16 + lm) * ATT_D + c * 32 + grp * 8]);
                cacc = __builtin_amdgcn_wmma_f32_16x16x32_f16(
                    false, a[c], false, bt, (short)0, cacc, false, false);
            }
            sc[n] = cacc;
        }

        // ---- key padding mask: one key per lane per tile (column = lm)
        #pragma unroll
        for (int n = 0; n < 4; ++n) {
            const bool msk = mask[(size_t)b * ATT_S + kb + n * 16 + lm] != 0;
            #pragma unroll
            for (int r = 0; r < 8; ++r)
                sc[n][r] = msk ? NEGBIG : sc[n][r];
        }

        // ---- online softmax (row stats live replicated across 16 lanes)
        float alpha[8];
        #pragma unroll
        for (int r = 0; r < 8; ++r) {
            float lmax = fmaxf(fmaxf(sc[0][r], sc[1][r]),
                               fmaxf(sc[2][r], sc[3][r]));
            float rm   = redmax16(lmax);
            float mnew = fmaxf(mrow[r], rm);
            alpha[r] = __expf(mrow[r] - mnew);
            mrow[r]  = mnew;
            float psum = 0.0f;
            #pragma unroll
            for (int n = 0; n < 4; ++n) {
                float p = __expf(sc[n][r] - mnew);
                sc[n][r] = p;
                psum += p;
                // P -> per-wave scratch, plain [row][key] f16
                Ps[wave][(r + 8 * grp) * BK + n * 16 + lm] = (_Float16)p;
            }
            lrow[r] = alpha[r] * lrow[r] + redsum16(psum);
            #pragma unroll
            for (int d = 0; d < 4; ++d) acc[d][r] *= alpha[r];
        }

        // make all lanes' P stores visible before cross-lane A-layout reads
        asm volatile("s_wait_dscnt 0" ::: "memory");

        // ---- reload P in A-matrix layout: two 16x32 chunks over the 64 keys
        v16h ap[2];
        #pragma unroll
        for (int c = 0; c < 2; ++c) {
            const _Float16* pp = &Ps[wave][lm * BK + c * 32];
            v8h plo = *(const v8h*)(pp + grp * 8);
            v8h phi = *(const v8h*)(pp + 16 + grp * 8);
            ap[c] = __builtin_shufflevector(plo, phi,
                0,1,2,3,4,5,6,7,8,9,10,11,12,13,14,15);
        }

        // ---- PV: 4 output d-chunks x 2 key-chunks (one WMMA each)
        // B element [key][d] = Vt[d*BK + key] -> contiguous per lane.
        #pragma unroll
        for (int dd = 0; dd < 4; ++dd) {
            #pragma unroll
            for (int c = 0; c < 2; ++c) {
                v16h bv = pack16(&Vt[(dd * 16 + lm) * BK + c * 32 + grp * 8]);
                acc[dd] = __builtin_amdgcn_wmma_f32_16x16x32_f16(
                    false, ap[c], false, bv, (short)0, acc[dd], false, false);
            }
        }

        __syncthreads();   // protect LDS K/V before next stage overwrites
    }

    // ---- epilogue: normalize and store fp32 output
    #pragma unroll
    for (int r = 0; r < 8; ++r) {
        const float inv = 1.0f / lrow[r];
        const size_t row = (size_t)bh * ATT_S + q0 + r + 8 * grp;
        #pragma unroll
        for (int dd = 0; dd < 4; ++dd)
            O[row * ATT_D + dd * 16 + lm] = acc[dd][r] * inv;
    }
}

extern "C" void kernel_launch(void* const* d_in, const int* in_sizes, int n_in,
                              void* d_out, int out_size, void* d_ws, size_t ws_size,
                              hipStream_t stream) {
    (void)in_sizes; (void)n_in; (void)out_size; (void)d_ws; (void)ws_size;
    const float*         q = (const float*)d_in[0];
    const float*         k = (const float*)d_in[1];
    const float*         v = (const float*)d_in[2];
    const unsigned char* m = (const unsigned char*)d_in[3];  // jax bool = 1 byte
    float* out = (float*)d_out;

    dim3 grid(ATT_S / BQ, ATT_B * ATT_H);   // (32, 64)
    sdpa_wmma_kernel<<<grid, 128, 0, stream>>>(q, k, v, m, out);
}
